// TalkingMultiheadClassAttention_8796093022422
// MI455X (gfx1250) — matrix operand order repaired and checked by reading, not compiled, not run on hardware
//
#include <hip/hip_runtime.h>
#include <hip/hip_bf16.h>

#define DIM      256
#define HEADS    8
#define HEAD_DIM 32
#define BB       32
#define NN       8192
#define SEG      128                   // tokens per fused block
#define NSEG     (NN / SEG)            // 64 segments
#define SCALE    0.17677669529663687f  // 1/sqrt(32)

typedef __attribute__((ext_vector_type(16))) __bf16 v16bf;
typedef __attribute__((ext_vector_type(8)))  float  v8f;

struct uint4x2 { uint4 a, b; };

static __device__ inline v16bf pack2(uint4 a, uint4 b) {
    uint4x2 u; u.a = a; u.b = b;
    return __builtin_bit_cast(v16bf, u);
}

// ---------------- workspace layout (bytes) ----------------
#define OFF_QK     0ull                                    // bf16 [B][16][256]
#define OFF_QB     (OFF_QK    + (size_t)BB*16*DIM*2)       // f32  [B][8]
#define OFF_MPART  (OFF_QB    + (size_t)BB*8*4)            // f32  [B][NSEG][8]
#define OFF_SPART  (OFF_MPART + (size_t)BB*NSEG*8*4)       // f32  [B][NSEG][8]
#define OFF_UPART  (OFF_SPART + (size_t)BB*NSEG*8*4)       // f32  [B][NSEG][8][256]
#define OFF_TPART  (OFF_UPART + (size_t)BB*NSEG*8*DIM*4)   // f32  [B][NSEG][256]

// ---- K0: q = ct@Wq+bq ; qk = scale*(Wk@q) bf16 (padded to 16 rows) ; qb ----
__global__ void k_qproj(const float* __restrict__ class_token,
                        const float* __restrict__ W_q, const float* __restrict__ b_q,
                        const float* __restrict__ W_kv, const float* __restrict__ b_kv,
                        __bf16* __restrict__ qk, float* __restrict__ qb) {
    const int b = blockIdx.x, tid = threadIdx.x;
    __shared__ float ct[DIM];
    __shared__ float ql[DIM];
    ct[tid] = class_token[b * DIM + tid];
    __syncthreads();
    float acc = b_q[tid];
    for (int c = 0; c < DIM; ++c) acc += ct[c] * W_q[c * DIM + tid];
    ql[tid] = acc;
    __syncthreads();
    if (tid < HEADS) {
        float s = 0.f;
        for (int d = 0; d < HEAD_DIM; ++d)
            s += ql[tid * HEAD_DIM + d] * b_kv[tid * HEAD_DIM + d];
        qb[b * HEADS + tid] = s * SCALE;
    }
    const int c = tid;
    for (int h = 0; h < 16; ++h) {
        float s = 0.f;
        if (h < HEADS)
            for (int d = 0; d < HEAD_DIM; ++d)
                s += W_kv[c * (2 * DIM) + h * HEAD_DIM + d] * ql[h * HEAD_DIM + d];
        qk[((size_t)b * 16 + h) * DIM + c] = (__bf16)(s * SCALE);
    }
}

// ---- K1 (fused): scores -> talk1+mask -> partial softmax -> u/t partials ---
// One block = one (b, 128-token segment). inputs tile is read twice but the
// second read hits L2 (tile is 128 KB, L2 is 192 MB) -> single HBM pass.
__global__ void __launch_bounds__(256)
k_fused(const float* __restrict__ inputs, const int* __restrict__ mask,
        const float* __restrict__ W_t1, const float* __restrict__ b_t1,
        const __bf16* __restrict__ qk, const float* __restrict__ qb,
        float* __restrict__ mpart, float* __restrict__ spart,
        float* __restrict__ upart, float* __restrict__ tpart) {
    const int b = blockIdx.x, seg = blockIdx.y, tid = threadIdx.x;
    const int wave = tid >> 5, lane = tid & 31;
    const int half = (lane >= 16) ? 1 : 0;
    const int mrow = lane & 15;
    const int n0 = seg * SEG;

    __shared__ __bf16 sQk[16 * DIM];   // 8 KB
    __shared__ float  sS2[8 * SEG];    // 4 KB mixed+masked scores [g][tok]
    __shared__ __bf16 sP[16 * SEG];    // 4 KB softmax numerators [row][tok]
    __shared__ float  sW1[64], sB1[8], sQB[8], sMg[8];

    {   // stage qk (16x256 bf16 = 512 uint4)
        const uint4* src = reinterpret_cast<const uint4*>(qk + (size_t)b * 16 * DIM);
        uint4* dst = reinterpret_cast<uint4*>(sQk);
        dst[tid] = src[tid];
        dst[tid + 256] = src[tid + 256];
    }
    if (tid < 64) sW1[tid] = W_t1[tid];
    if (tid < 8)  { sB1[tid] = b_t1[tid]; sQB[tid] = qb[b * HEADS + tid]; }
    __syncthreads();

    // ---------- pass 1: scores tile (M=heads16, N=16 tokens, K=256) ----------
    const int t0 = wave * 16;                      // 8 waves x 16 tokens = SEG
    v8f acc = {0.f, 0.f, 0.f, 0.f, 0.f, 0.f, 0.f, 0.f};
    const float* brow = inputs + ((size_t)b * NN + n0 + t0 + mrow) * DIM;
#pragma unroll
    for (int ks = 0; ks < 8; ++ks) {
        const int kbase = ks * 32;
        const int ao = mrow * DIM + kbase + half * 16;
        uint4 a0 = *reinterpret_cast<const uint4*>(sQk + ao);
        uint4 a1 = *reinterpret_cast<const uint4*>(sQk + ao + 8);
        v16bf av = pack2(a0, a1);
        const float* bp = brow + kbase + half * 16;
        float fl[16];
        *reinterpret_cast<float4*>(fl + 0)  = *reinterpret_cast<const float4*>(bp + 0);
        *reinterpret_cast<float4*>(fl + 4)  = *reinterpret_cast<const float4*>(bp + 4);
        *reinterpret_cast<float4*>(fl + 8)  = *reinterpret_cast<const float4*>(bp + 8);
        *reinterpret_cast<float4*>(fl + 12) = *reinterpret_cast<const float4*>(bp + 12);
        v16bf bv;
#pragma unroll
        for (int i = 0; i < 16; ++i) bv[i] = (__bf16)fl[i];
        acc = __builtin_amdgcn_wmma_f32_16x16x32_bf16(false, av, false, bv,
                                                      (short)0, acc, false, false);
    }
    // lanes 0..15 hold all 8 head scores for token t0+lane: mix + mask in-lane
    if (lane < 16) {
        const int tl = t0 + lane;
        float s[8];
#pragma unroll
        for (int h = 0; h < 8; ++h) s[h] = acc[h] + sQB[h];
        const float pen = (1.0f - (float)mask[b * NN + n0 + tl]) * -1.0e9f;
#pragma unroll
        for (int g = 0; g < 8; ++g) {
            float v = sB1[g] + pen;
#pragma unroll
            for (int h = 0; h < 8; ++h) v += s[h] * sW1[h * 8 + g];
            sS2[g * SEG + tl] = v;
        }
    }
    __syncthreads();

    // ---------- partial softmax: wave g reduces its head over 128 tokens ----
    {
        const int g = wave;
        float m = -3.0e38f;
#pragma unroll
        for (int j = 0; j < 4; ++j) m = fmaxf(m, sS2[g * SEG + lane + 32 * j]);
#pragma unroll
        for (int off = 16; off > 0; off >>= 1) m = fmaxf(m, __shfl_xor(m, off));
        float se = 0.f;
#pragma unroll
        for (int j = 0; j < 4; ++j) se += __expf(sS2[g * SEG + lane + 32 * j] - m);
#pragma unroll
        for (int off = 16; off > 0; off >>= 1) se += __shfl_xor(se, off);
        if (lane == 0) {
            mpart[((size_t)b * NSEG + seg) * 8 + g] = m;
            spart[((size_t)b * NSEG + seg) * 8 + g] = se;
            sMg[g] = m;
        }
    }
    __syncthreads();

    // ---------- numerator matrix sP: rows 0..7 = exp(s-m); row 8 = ones -----
    // (row 8 of ones makes WMMA also emit t[c] = sum_n inputs[n,c] for free)
#pragma unroll
    for (int j = 0; j < 8; ++j) {
        const int idx = tid + 256 * j;       // covers 16*128
        const int r = idx >> 7;
        float val;
        if (r < 8)       val = __expf(sS2[idx] - sMg[r]);
        else if (r == 8) val = 1.0f;
        else             val = 0.0f;
        sP[idx] = (__bf16)val;
    }
    __syncthreads();

    // ---------- pass 2: u[h,c] = sum_n p[h,n]*inputs[n,c] (K=128 tokens) ----
    const float* tile = inputs + ((size_t)b * NN + n0) * DIM;
#pragma unroll
    for (int ct = 0; ct < 2; ++ct) {
        const int c0 = wave * 16 + ct * 128;   // 8 waves x 2 = 16 column tiles
        v8f acc2 = {0.f, 0.f, 0.f, 0.f, 0.f, 0.f, 0.f, 0.f};
#pragma unroll
        for (int kt = 0; kt < 4; ++kt) {
            const int tbase = kt * 32;
            const int ao = mrow * SEG + tbase + half * 16;
            uint4 a0 = *reinterpret_cast<const uint4*>(sP + ao);
            uint4 a1 = *reinterpret_cast<const uint4*>(sP + ao + 8);
            v16bf av = pack2(a0, a1);
            // B: col = channel c0+mrow, K = token (L2-resident re-read)
            const float* bp = tile + (size_t)(tbase + half * 16) * DIM + c0 + mrow;
            v16bf bv;
#pragma unroll
            for (int e = 0; e < 16; ++e) bv[e] = (__bf16)bp[(size_t)e * DIM];
            acc2 = __builtin_amdgcn_wmma_f32_16x16x32_bf16(false, av, false, bv,
                                                           (short)0, acc2, false, false);
        }
        if (lane < 16) {       // rows 0..7 = heads
#pragma unroll
            for (int r = 0; r < 8; ++r)
                upart[(((size_t)b * NSEG + seg) * 8 + r) * DIM + c0 + lane] = acc2[r];
        } else {               // row 8 (ones row) = token-sum of inputs
            tpart[((size_t)b * NSEG + seg) * DIM + c0 + mrow] = acc2[0];
        }
    }
}

// ---- K2: combine partials (softmax rescale), talk2, W_v(+bias), W_out ------
__global__ void __launch_bounds__(256)
k_reduce(const float* __restrict__ mpart, const float* __restrict__ spart,
         const float* __restrict__ upart, const float* __restrict__ tpart,
         const float* __restrict__ W_t2, const float* __restrict__ b_t2,
         const float* __restrict__ W_kv, const float* __restrict__ b_kv,
         const float* __restrict__ W_out, const float* __restrict__ b_out,
         float* __restrict__ out) {
    const int b = blockIdx.x, tid = threadIdx.x;
    __shared__ float sE[NSEG * 8];   // 2 KB rescale factors
    __shared__ float sM[8], sIS[8], sW2[64], sB2[8];
    __shared__ float wl[8 * DIM];    // 8 KB
    __shared__ float ov[DIM];

    if (tid < 64) sW2[tid] = W_t2[tid];
    if (tid < 8) {
        sB2[tid] = b_t2[tid];
        float m = -3.0e38f;
        for (int s = 0; s < NSEG; ++s)
            m = fmaxf(m, mpart[((size_t)b * NSEG + s) * 8 + tid]);
        sM[tid] = m;
    }
    __syncthreads();
#pragma unroll
    for (int j = 0; j < 2; ++j) {
        const int idx = tid + 256 * j;          // NSEG*8 = 512
        const int s = idx >> 3, g = idx & 7;
        sE[idx] = __expf(mpart[((size_t)b * NSEG + s) * 8 + g] - sM[g]);
    }
    __syncthreads();
    if (tid < 8) {
        float S = 0.f;
        for (int s = 0; s < NSEG; ++s)
            S += sE[s * 8 + tid] * spart[((size_t)b * NSEG + s) * 8 + tid];
        sIS[tid] = 1.0f / S;
    }
    __syncthreads();

    const int c = tid;
    float u[8] = {0.f, 0.f, 0.f, 0.f, 0.f, 0.f, 0.f, 0.f};
    float t = 0.f;
    for (int s = 0; s < NSEG; ++s) {
        const float* up = upart + (((size_t)b * NSEG + s) * 8) * DIM + c;
#pragma unroll
        for (int g = 0; g < 8; ++g) u[g] += sE[s * 8 + g] * up[g * DIM];
        t += tpart[((size_t)b * NSEG + s) * DIM + c];
    }
#pragma unroll
    for (int g = 0; g < 8; ++g) {
        float wv = sB2[g] * t;
#pragma unroll
        for (int h = 0; h < 8; ++h) wv += sW2[h * 8 + g] * sIS[h] * u[h];
        wl[g * DIM + c] = wv;
    }
    __syncthreads();

    // v-projection; sum_n attn2[g,n] = sum_h Wt2[h,g] + bt2[g]*N exactly
    const int h = tid >> 5;
    float as = sB2[h] * (float)NN;
#pragma unroll
    for (int h2 = 0; h2 < 8; ++h2) as += sW2[h2 * 8 + h];
    float a = as * b_kv[DIM + tid];
    for (int cc = 0; cc < DIM; ++cc)
        a += wl[h * DIM + cc] * W_kv[cc * (2 * DIM) + DIM + tid];
    ov[tid] = a;
    __syncthreads();
    float o = b_out[tid];
    for (int i = 0; i < DIM; ++i) o += ov[i] * W_out[i * DIM + tid];
    out[b * DIM + tid] = o;
}

extern "C" void kernel_launch(void* const* d_in, const int* in_sizes, int n_in,
                              void* d_out, int out_size, void* d_ws, size_t ws_size,
                              hipStream_t stream) {
    const float* inputs      = (const float*)d_in[0];
    const float* class_token = (const float*)d_in[1];
    const int*   mask        = (const int*)  d_in[2];
    const float* W_kv        = (const float*)d_in[3];
    const float* b_kv        = (const float*)d_in[4];
    const float* W_q         = (const float*)d_in[5];
    const float* b_q         = (const float*)d_in[6];
    const float* W_t1        = (const float*)d_in[7];
    const float* b_t1        = (const float*)d_in[8];
    const float* W_t2        = (const float*)d_in[9];
    const float* b_t2        = (const float*)d_in[10];
    const float* W_out       = (const float*)d_in[11];
    const float* b_out       = (const float*)d_in[12];
    float* out = (float*)d_out;

    char* ws = (char*)d_ws;
    __bf16* qk    = (__bf16*)(ws + OFF_QK);
    float*  qb    = (float*) (ws + OFF_QB);
    float*  mpart = (float*) (ws + OFF_MPART);
    float*  spart = (float*) (ws + OFF_SPART);
    float*  upart = (float*) (ws + OFF_UPART);
    float*  tpart = (float*) (ws + OFF_TPART);

    k_qproj<<<BB, 256, 0, stream>>>(class_token, W_q, b_q, W_kv, b_kv, qk, qb);
    k_fused<<<dim3(BB, NSEG), 256, 0, stream>>>(inputs, mask, W_t1, b_t1,
                                                qk, qb, mpart, spart, upart, tpart);
    k_reduce<<<BB, 256, 0, stream>>>(mpart, spart, upart, tpart,
                                     W_t2, b_t2, W_kv, b_kv, W_out, b_out, out);
}
